// HTQNet_60868276519088
// MI455X (gfx1250) — matrix operand order repaired and checked by reading, not compiled, not run on hardware
//
#include <hip/hip_runtime.h>

#define HH 256
#define INF 512
#define BATCH 32
#define MAXP 10
#define MAXV 150
#define NB 16                 // persistent blocks
#define HS (HH / NB)          // 16 hidden units per block
#define WPITCH 264            // halves; 264*2B=528B => DW stride 132, 132%64=4 -> conflict-free B loads
#define GPITCH 68             // gates LDS pitch (floats), padded

typedef __attribute__((ext_vector_type(16))) _Float16 v16h;
typedef __attribute__((ext_vector_type(8)))  _Float16 v8h;
typedef __attribute__((ext_vector_type(8)))  float    v8f;

__device__ __forceinline__ float sigm(float x) { return 1.0f / (1.0f + __expf(-x)); }

// ---- WMMA fragment loaders -------------------------------------------------
// A: 16x32 f16 tile from act[32][256] row-major (global, f16).
// Layout (ISA 7.12.2): lane<16 -> M=lane, K 0..7 & 16..23 ; lane>=16 -> K 8..15 & 24..31
__device__ __forceinline__ v16h loadA(const _Float16* base, int mTile, int lane, int kBlock) {
  int r  = mTile * 16 + (lane & 15);
  int kb = kBlock * 32 + ((lane >> 4) << 3);
  const _Float16* p = base + r * HH + kb;
  union { v16h v; v8h h[2]; } u;
  u.h[0] = *(const v8h*)(p);
  u.h[1] = *(const v8h*)(p + 16);
  return u.v;
}
// B: 32x16 f16 tile; weights stored [n][k] (so B[k][n] = w[n][k]).
// Layout: lane<16 -> N=lane, K 0..7 & 16..23 ; lane>=16 -> N=lane-16, K 8..15 & 24..31
__device__ __forceinline__ v16h loadB(const _Float16* wtile, int lane, int kBlock) {
  int n  = lane & 15;
  int kb = kBlock * 32 + ((lane >> 4) << 3);
  const _Float16* p = wtile + n * WPITCH + kb;
  union { v16h v; v8h h[2]; } u;
  u.h[0] = *(const v8h*)(p);
  u.h[1] = *(const v8h*)(p + 16);
  return u.v;
}

__device__ __forceinline__ v8f gemm8(v8f acc, const _Float16* A, const _Float16* Wt,
                                     int mTile, int lane) {
#pragma unroll
  for (int k = 0; k < 8; ++k) {
    v16h a = loadA(A, mTile, lane, k);
    v16h b = loadB(Wt, lane, k);
    acc = __builtin_amdgcn_wmma_f32_16x16x32_f16(false, a, false, b, (short)0, acc,
                                                 false, false);
  }
  return acc;
}

// ---- grid barrier (gen-counter, L2 atomics, s_sleep spin) ------------------
__device__ __forceinline__ void grid_barrier(int* cnt, int* gen) {
  __syncthreads();
  if (threadIdx.x == 0) {
    __threadfence();
    int g = __hip_atomic_load(gen, __ATOMIC_RELAXED, __HIP_MEMORY_SCOPE_AGENT);
    int arrived = __hip_atomic_fetch_add(cnt, 1, __ATOMIC_ACQ_REL, __HIP_MEMORY_SCOPE_AGENT);
    if (arrived == NB - 1) {
      __hip_atomic_store(cnt, 0, __ATOMIC_RELAXED, __HIP_MEMORY_SCOPE_AGENT);
      __hip_atomic_fetch_add(gen, 1, __ATOMIC_RELEASE, __HIP_MEMORY_SCOPE_AGENT);
    } else {
      while (__hip_atomic_load(gen, __ATOMIC_ACQUIRE, __HIP_MEMORY_SCOPE_AGENT) == g)
        __builtin_amdgcn_s_sleep(2);
    }
  }
  __syncthreads();
  __threadfence();
}

// ---- prep kernels ----------------------------------------------------------
// W_comb[cg][k] = sum_m w_ih0[cg][m] * w_h2i[m][k]   (fold autoregressive input)
__global__ void k_fold_wcomb(const float* __restrict__ wih0, const float* __restrict__ wh2i,
                             _Float16* __restrict__ wc) {
  __shared__ float rowbuf[INF];
  int cg = blockIdx.x;
  for (int m = threadIdx.x; m < INF; m += 256) rowbuf[m] = wih0[cg * INF + m];
  __syncthreads();
  int k = threadIdx.x;
  float s = 0.f;
  for (int m = 0; m < INF; ++m) s = fmaf(rowbuf[m], wh2i[m * HH + k], s);
  wc[cg * HH + k] = (_Float16)s;
}

__global__ void k_convert3(const float* a, const float* b, const float* c,
                           _Float16* oa, _Float16* ob, _Float16* oc) {
  int i = blockIdx.x * 256 + threadIdx.x;  // 1024*256 = 262144 elements exactly
  oa[i] = (_Float16)a[i];
  ob[i] = (_Float16)b[i];
  oc[i] = (_Float16)c[i];
}

__global__ void k_bias(const float* bih0, const float* bhh0, const float* bih1,
                       const float* bhh1, const float* wih0, const float* bh2i,
                       float* b0, float* b1, float* bx) {
  int i = blockIdx.x * 256 + threadIdx.x;  // 0..1023
  b0[i] = bih0[i] + bhh0[i];
  b1[i] = bih1[i] + bhh1[i];
  float s = 0.f;
  for (int m = 0; m < INF; ++m) s = fmaf(wih0[i * INF + m], bh2i[m], s);
  bx[i] = s;
}

__global__ void k_g0(const float* __restrict__ feat, const float* __restrict__ wih0,
                     float* __restrict__ g0) {
  __shared__ float f[INF];
  int b = blockIdx.x;
  for (int m = threadIdx.x; m < INF; m += 256) f[m] = feat[b * INF + m];
  __syncthreads();
  for (int cg = threadIdx.x; cg < 4 * HH; cg += 256) {
    float s = 0.f;
    for (int m = 0; m < INF; ++m) s = fmaf(f[m], wih0[cg * INF + m], s);
    g0[b * 4 * HH + cg] = s;
  }
}

__global__ void k_zero(_Float16* h0, _Float16* h1, int* cnt, int* gen) {
  int i = blockIdx.x * 256 + threadIdx.x;
  int stride = gridDim.x * 256;
  for (int j = i; j < 2 * BATCH * HH; j += stride) { h0[j] = (_Float16)0.f; h1[j] = (_Float16)0.f; }
  if (i == 0) { *cnt = 0; *gen = 0; }
}

// ---- persistent LSTM kernel ------------------------------------------------
__global__ __launch_bounds__(256) void lstm_persistent(
    const _Float16* __restrict__ Wc, const _Float16* __restrict__ Whh0,
    const _Float16* __restrict__ Wih1, const _Float16* __restrict__ Whh1,
    const float* __restrict__ b0, const float* __restrict__ b1,
    const float* __restrict__ bx, const float* __restrict__ G0,
    _Float16* __restrict__ h0buf, _Float16* __restrict__ h1buf,
    const float* __restrict__ wv, const float* __restrict__ bv,
    float* __restrict__ out, int* cnt, int* gen) {
  __shared__ _Float16 lWc[64 * WPITCH];
  __shared__ _Float16 lWhh0[64 * WPITCH];
  __shared__ _Float16 lWih1[64 * WPITCH];
  __shared__ _Float16 lWhh1[64 * WPITCH];
  __shared__ float lG[32 * GPITCH];
  __shared__ float c0[512];
  __shared__ float c1[512];

  const int bid  = blockIdx.x;
  const int tid  = threadIdx.x;
  const int lane = tid & 31;
  const int wave = tid >> 5;
  const int mTile = wave & 1;   // batch tile 0/1
  const int gate  = wave >> 1;  // 0=i 1=f 2=g 3=o  (16 gate columns each)

  // stage this block's weight slices into LDS (once, persistent)
  auto stage = [&](const _Float16* src, _Float16* dst) {
    for (int idx = tid; idx < 64 * 16; idx += 256) {
      int n = idx >> 4, seg = idx & 15;
      int grow = (n >> 4) * HH + bid * HS + (n & 15);  // global gate-row
      *(v8h*)(dst + n * WPITCH + seg * 16)     = *(const v8h*)(src + grow * HH + seg * 16);
      *(v8h*)(dst + n * WPITCH + seg * 16 + 8) = *(const v8h*)(src + grow * HH + seg * 16 + 8);
    }
  };
  stage(Wc, lWc); stage(Whh0, lWhh0); stage(Wih1, lWih1); stage(Whh1, lWhh1);
  for (int i = tid; i < 512; i += 256) { c0[i] = 0.f; c1[i] = 0.f; }
  __syncthreads();

  for (int s = 0; s < MAXP * MAXV; ++s) {
    int p = s / MAXV, v = s % MAXV;
    int rp = s & 1, wp = rp ^ 1;
    const _Float16* h0r = h0buf + rp * BATCH * HH;
    _Float16*       h0w = h0buf + wp * BATCH * HH;
    const _Float16* h1r = h1buf + rp * BATCH * HH;
    _Float16*       h1w = h1buf + wp * BATCH * HH;

    // ---- layer 0: gates = (v>0 ? h1_prev@Wc : G0) + h0_prev@Whh0 + b0 (+bx)
    {
      v8f acc = {};
      if (v > 0) acc = gemm8(acc, h1r, lWc + gate * 16 * WPITCH, mTile, lane);
      acc = gemm8(acc, h0r, lWhh0 + gate * 16 * WPITCH, mTile, lane);
#pragma unroll
      for (int e = 0; e < 8; ++e)
        lG[(mTile * 16 + e + ((lane >> 4) << 3)) * GPITCH + gate * 16 + (lane & 15)] = acc[e];
    }
    __syncthreads();
    for (int u = tid; u < 512; u += 256) {
      int row = u >> 4, jl = u & 15, j = bid * HS + jl;
      float ei, ef, eg, eo;
      if (v == 0) { const float* g = G0 + row * 4 * HH;
        ei = g[j]; ef = g[256 + j]; eg = g[512 + j]; eo = g[768 + j];
      } else { ei = bx[j]; ef = bx[256 + j]; eg = bx[512 + j]; eo = bx[768 + j]; }
      float xi = lG[row * GPITCH + jl]      + b0[j]       + ei;
      float xf = lG[row * GPITCH + 16 + jl] + b0[256 + j] + ef;
      float xg = lG[row * GPITCH + 32 + jl] + b0[512 + j] + eg;
      float xo = lG[row * GPITCH + 48 + jl] + b0[768 + j] + eo;
      float cn = sigm(xf) * c0[u] + sigm(xi) * tanhf(xg);
      c0[u] = cn;
      h0w[row * HH + j] = (_Float16)(sigm(xo) * tanhf(cn));
    }
    grid_barrier(cnt, gen);

    // ---- layer 1: gates = h0_new@Wih1 + h1_prev@Whh1 + b1
    {
      v8f acc = {};
      acc = gemm8(acc, h0w, lWih1 + gate * 16 * WPITCH, mTile, lane);
      acc = gemm8(acc, h1r, lWhh1 + gate * 16 * WPITCH, mTile, lane);
#pragma unroll
      for (int e = 0; e < 8; ++e)
        lG[(mTile * 16 + e + ((lane >> 4) << 3)) * GPITCH + gate * 16 + (lane & 15)] = acc[e];
    }
    __syncthreads();
    for (int u = tid; u < 512; u += 256) {
      int row = u >> 4, jl = u & 15, j = bid * HS + jl;
      float xi = lG[row * GPITCH + jl]      + b1[j];
      float xf = lG[row * GPITCH + 16 + jl] + b1[256 + j];
      float xg = lG[row * GPITCH + 32 + jl] + b1[512 + j];
      float xo = lG[row * GPITCH + 48 + jl] + b1[768 + j];
      float cn = sigm(xf) * c1[u] + sigm(xi) * tanhf(xg);
      c1[u] = cn;
      h1w[row * HH + j] = (_Float16)(sigm(xo) * tanhf(cn));
    }
    grid_barrier(cnt, gen);

    // ---- vertex = h1_new @ w_v.T + b_v ; each block emits 4 of 64 outputs
    if (wave < 4) {
      int o = bid * 4 + wave;
      int brow = o >> 1, col = o & 1;
      const float*    wr = wv + col * HH;
      const _Float16* hr = h1w + brow * HH;
      float partial = 0.f;
      int k0 = lane * 8;
#pragma unroll
      for (int k = 0; k < 8; ++k) partial = fmaf((float)hr[k0 + k], wr[k0 + k], partial);
      for (int off = 16; off > 0; off >>= 1) partial += __shfl_down(partial, off, 32);
      if (lane == 0)
        out[((brow * MAXP + p) * MAXV + v) * 2 + col] = partial + bv[col];
    }
  }
}

// ---- host launcher ---------------------------------------------------------
extern "C" void kernel_launch(void* const* d_in, const int* in_sizes, int n_in,
                              void* d_out, int out_size, void* d_ws, size_t ws_size,
                              hipStream_t stream) {
  (void)in_sizes; (void)n_in; (void)out_size; (void)ws_size;
  const float* feat = (const float*)d_in[0];
  const float* wih0 = (const float*)d_in[1];
  const float* whh0 = (const float*)d_in[2];
  const float* bih0 = (const float*)d_in[3];
  const float* bhh0 = (const float*)d_in[4];
  const float* wih1 = (const float*)d_in[5];
  const float* whh1 = (const float*)d_in[6];
  const float* bih1 = (const float*)d_in[7];
  const float* bhh1 = (const float*)d_in[8];
  const float* wv   = (const float*)d_in[9];
  const float* bv   = (const float*)d_in[10];
  const float* wh2i = (const float*)d_in[11];
  const float* bh2i = (const float*)d_in[12];
  float* out = (float*)d_out;

  char* ws = (char*)d_ws;
  size_t off = 0;
  auto take = [&](size_t bytes) -> char* {
    char* p = ws + off;
    off = (off + bytes + 255) & ~(size_t)255;
    return p;
  };
  int* cnt = (int*)take(128);
  int* gen = (int*)take(128);
  _Float16* h0buf = (_Float16*)take(2 * BATCH * HH * sizeof(_Float16));
  _Float16* h1buf = (_Float16*)take(2 * BATCH * HH * sizeof(_Float16));
  _Float16* Wc16    = (_Float16*)take((size_t)4 * HH * HH * sizeof(_Float16));
  _Float16* Whh0_16 = (_Float16*)take((size_t)4 * HH * HH * sizeof(_Float16));
  _Float16* Wih1_16 = (_Float16*)take((size_t)4 * HH * HH * sizeof(_Float16));
  _Float16* Whh1_16 = (_Float16*)take((size_t)4 * HH * HH * sizeof(_Float16));
  float* b0 = (float*)take(4 * HH * sizeof(float));
  float* b1 = (float*)take(4 * HH * sizeof(float));
  float* bx = (float*)take(4 * HH * sizeof(float));
  float* G0 = (float*)take((size_t)BATCH * 4 * HH * sizeof(float));

  k_fold_wcomb<<<dim3(4 * HH), dim3(256), 0, stream>>>(wih0, wh2i, Wc16);
  k_convert3<<<dim3(1024), dim3(256), 0, stream>>>(whh0, wih1, whh1,
                                                   Whh0_16, Wih1_16, Whh1_16);
  k_bias<<<dim3(4), dim3(256), 0, stream>>>(bih0, bhh0, bih1, bhh1, wih0, bh2i, b0, b1, bx);
  k_g0<<<dim3(BATCH), dim3(256), 0, stream>>>(feat, wih0, G0);
  k_zero<<<dim3(32), dim3(256), 0, stream>>>(h0buf, h1buf, cnt, gen);
  lstm_persistent<<<dim3(NB), dim3(256), 0, stream>>>(
      Wc16, Whh0_16, Wih1_16, Whh1_16, b0, b1, bx, G0, h0buf, h1buf, wv, bv, out, cnt, gen);
}
